// CrossAttentionLayer_111669150277
// MI455X (gfx1250) — matrix operand order repaired and checked by reading, not compiled, not run on hardware
//
#include <hip/hip_runtime.h>
#include <hip/hip_bf16.h>
#include <math.h>
#include <stdint.h>

// ---------------------------------------------------------------------------
// CrossAttentionLayer for MI455X (gfx1250, wave32, WMMA + async-to-LDS).
// B=2 S=2048 K=2048 D=1024 H=16 HD=64 F=4096.  bf16 WMMA, f32 accumulate.
// ---------------------------------------------------------------------------

typedef __bf16 bf16;
typedef __attribute__((ext_vector_type(16))) __bf16 v16bf;
typedef __attribute__((ext_vector_type(8)))  __bf16 bf16x8;
typedef __attribute__((ext_vector_type(8)))  float  v8f;

// ---- WMMA wrapper: D = A(16x32 bf16) * B(32x16 bf16) + C(16x16 f32) -------
__device__ __forceinline__ v8f wmma_bf16(v16bf a, v16bf b, v8f c) {
  return __builtin_amdgcn_wmma_f32_16x16x32_bf16(
      /*neg_a=*/false, a, /*neg_b=*/false, b,
      /*c_mod=*/(short)0, c, /*reuse_a=*/false, /*reuse_b=*/false);
}

// ---- A fragment (16x32, MxK), source row-major [16, >=32], stride ld ------
// lanes 0-15: row=lane, K=[0..7]&[16..23]; lanes 16-31: row=lane-16,
// K=[8..15]&[24..31].  Two 16B loads per lane.
__device__ __forceinline__ v16bf load_a_frag(const bf16* __restrict__ base,
                                             int ld, int lane) {
  const int row  = lane & 15;
  const int kofs = (lane >> 4) * 8;
  const bf16* p = base + (size_t)row * ld + kofs;
  bf16x8 lo = *(const bf16x8*)(p);
  bf16x8 hi = *(const bf16x8*)(p + 16);
  v16bf r;
#pragma unroll
  for (int i = 0; i < 8; ++i) { r[i] = lo[i]; r[i + 8] = hi[i]; }
  return r;
}

// ---- B fragment (32x16, KxN): lane holds column n=lane&15, K=[kofs..kofs+7]
// & [kofs+16..kofs+23], kofs=(lane>>4)*8.  Source = [N,K] row-major rows (weights / K / Vt).
__device__ __forceinline__ v16bf load_b_frag(const bf16* __restrict__ base,
                                             int ld, int lane) {
  const int n    = lane & 15;
  const int kofs = (lane >> 4) * 8;
  const bf16* p = base + (size_t)n * ld + kofs;
  bf16x8 lo = *(const bf16x8*)(p);
  bf16x8 hi = *(const bf16x8*)(p + 16);
  v16bf r;
#pragma unroll
  for (int i = 0; i < 8; ++i) { r[i] = lo[i]; r[i + 8] = hi[i]; }
  return r;
}

// ---- CDNA5 async copy: 16B global -> LDS, ASYNCcnt-tracked ----------------
__device__ __forceinline__ void async_g2l_b128(uint32_t lds_off,
                                               const void* gaddr) {
  asm volatile("global_load_async_to_lds_b128 %0, %1, off"
               :: "v"(lds_off), "v"((uint64_t)(uintptr_t)gaddr)
               : "memory");
}
__device__ __forceinline__ void wait_asynccnt0() {
  asm volatile("s_wait_asynccnt 0x0" ::: "memory");
}

// ---------------------------------------------------------------------------
// f32 -> bf16 conversion (grid-stride)
// ---------------------------------------------------------------------------
__global__ void cvt_bf16(const float* __restrict__ in, bf16* __restrict__ out,
                         int n) {
  int i = blockIdx.x * blockDim.x + threadIdx.x;
  const int stride = gridDim.x * blockDim.x;
  for (; i < n; i += stride) out[i] = (bf16)in[i];
}

// ---------------------------------------------------------------------------
// WMMA GEMM:  C[M,N] = A[M,K] * Wt[N,K]^T + bias[N]
// Block = 256 threads = 8 waves stacked along M -> 128x64 tile.
// B tile (64 rows x 32 K, 4KB) staged in LDS via global_load_async_to_lds,
// double buffered; A fragment register double-buffered.
// mode 0: Cf f32 row-major
// mode 1: GELU(exact) -> Cb bf16 row-major
// mode 2: Q/K head store : Cb[((b*16+h)*L + l)*64 + hd]    (bf16)
// mode 3: V transposed   : Cb[((b*16+h)*64 + hd)*L + l]    (bf16)
// ---------------------------------------------------------------------------
__global__ __launch_bounds__(256) void gemm_bf16_wmma(
    const bf16* __restrict__ A, const bf16* __restrict__ Wt,
    const float* __restrict__ bias, float* __restrict__ Cf,
    bf16* __restrict__ Cb, int M, int N, int Kd, int mode, int L) {
  __shared__ __align__(16) bf16 bsm[2][64 * 32];
  const int lane = threadIdx.x & 31;
  const int wave = threadIdx.x >> 5;
  const int m0 = blockIdx.y * 128 + wave * 16;
  const int n0 = blockIdx.x * 64;
  const int tid = threadIdx.x;

  // Cooperative B-tile fetch: thread t -> row n = t>>2, 16B chunk (t&3)*8.
  const int brow = tid >> 2;
  const int bk8  = (tid & 3) * 8;
  const bf16* bsrc = Wt + (size_t)(n0 + brow) * Kd + bk8;
  const uint32_t lds0 = (uint32_t)(uintptr_t)(&bsm[0][brow * 32 + bk8]);
  const uint32_t lds1 = (uint32_t)(uintptr_t)(&bsm[1][brow * 32 + bk8]);

  // issue first async tile (k = 0) into buffer 0
  async_g2l_b128(lds0, bsrc);

  const bf16* Ab = A + (size_t)m0 * Kd;
  v8f acc[4] = {};
  v16bf af = load_a_frag(Ab, Kd, lane);

  const int nIter = Kd >> 5;
  for (int it = 0; it < nIter; ++it) {
    const int cur = it & 1;
    wait_asynccnt0();   // my slice of tile `it` has landed in LDS
    __syncthreads();    // everyone's slice landed; tile it-1 fully consumed

    // stream tile it+1 into the other buffer while we compute
    if (it + 1 < nIter)
      async_g2l_b128(cur ? lds0 : lds1, bsrc + (size_t)(it + 1) * 32);

    // register double-buffer the A fragment
    v16bf afn = af;
    if (it + 1 < nIter) afn = load_a_frag(Ab + (it + 1) * 32, Kd, lane);
    __builtin_prefetch(Ab + (it + 4) * 32, 0, 3);  // global_prefetch_b8

    const bf16* bs = &bsm[cur][0];
    v16bf b0 = load_b_frag(bs + 0 * 16 * 32, 32, lane);
    v16bf b1 = load_b_frag(bs + 1 * 16 * 32, 32, lane);
    v16bf b2 = load_b_frag(bs + 2 * 16 * 32, 32, lane);
    v16bf b3 = load_b_frag(bs + 3 * 16 * 32, 32, lane);
    acc[0] = wmma_bf16(af, b0, acc[0]);
    acc[1] = wmma_bf16(af, b1, acc[1]);
    acc[2] = wmma_bf16(af, b2, acc[2]);
    acc[3] = wmma_bf16(af, b3, acc[3]);
    af = afn;
  }

  // C layout: component r on lanes 0-15 -> M=m0+r, lanes 16-31 -> M=m0+r+8.
  const int col = lane & 15, half = lane >> 4;
#pragma unroll
  for (int t = 0; t < 4; ++t) {
    const int n = n0 + t * 16 + col;
    const float bv = bias[n];
#pragma unroll
    for (int r = 0; r < 8; ++r) {
      const int m = m0 + r + half * 8;
      float v = acc[t][r] + bv;
      if (mode == 0) {
        Cf[(size_t)m * N + n] = v;
      } else if (mode == 1) {
        v = 0.5f * v * (1.0f + erff(v * 0.70710678118654752f));  // exact GELU
        Cb[(size_t)m * N + n] = (bf16)v;
      } else {
        const int b = m / L, l = m - b * L;
        const int h = n >> 6, hd = n & 63;
        if (mode == 2)
          Cb[(((size_t)(b * 16 + h)) * L + l) * 64 + hd] = (bf16)v;
        else
          Cb[(((size_t)(b * 16 + h)) * 64 + hd) * L + l] = (bf16)v;
      }
    }
  }
}

// ---------------------------------------------------------------------------
// Flash-style attention, one wave per 16-query tile.
// Q : [B,H,2048,64] bf16,  Km : [B,H,2048,64] bf16,  Vt : [B,H,64,2048] bf16
// O : [B,H,2048,64] f32 (raw flat order; LN kernel consumes it flat).
// Per 32-kv step: 4 WMMA QK^T, online softmax (__expf + shfl reductions),
// P routed through LDS (C-layout -> A-layout), 4 WMMA P*V.  V fragments are
// fetched before the softmax so they overlap the LDS round-trip.
// ---------------------------------------------------------------------------
#define ATTN_WPB 4
__global__ __launch_bounds__(ATTN_WPB * 32) void attn_wmma(
    const bf16* __restrict__ Q, const bf16* __restrict__ Km,
    const bf16* __restrict__ Vt, float* __restrict__ O) {
  __shared__ __align__(16) bf16 psmem[ATTN_WPB][16 * 32];
  const int lane = threadIdx.x & 31;
  const int wave = threadIdx.x >> 5;
  const int b = blockIdx.z, h = blockIdx.y;
  const int q0 = (blockIdx.x * ATTN_WPB + wave) * 16;
  const size_t bh = (size_t)(b * 16 + h);

  const bf16* qb = Q + (bh * 2048 + q0) * 64;
  const bf16* kb = Km + bh * 2048 * 64;
  const bf16* vb = Vt + bh * 64 * 2048;

  // Q fragments stay resident for the whole KV sweep (HD=64 -> 2 A-frags).
  const v16bf qf0 = load_a_frag(qb, 64, lane);
  const v16bf qf1 = load_a_frag(qb + 32, 64, lane);

  v8f o[4] = {};
  float mrow[8], lrow[8];
#pragma unroll
  for (int r = 0; r < 8; ++r) { mrow[r] = -3.0e38f; lrow[r] = 0.f; }

  bf16* ps = &psmem[wave][0];
  const int col = lane & 15, half = lane >> 4;
  const float scale = 0.03125f;  // 1/sqrt(D) = 1/32 (reference scales by D!)

  for (int j = 0; j < 2048; j += 32) {
    v8f s0 = {}, s1 = {};
    s0 = wmma_bf16(qf0, load_b_frag(kb + (size_t)j * 64, 64, lane), s0);
    s0 = wmma_bf16(qf1, load_b_frag(kb + (size_t)j * 64 + 32, 64, lane), s0);
    s1 = wmma_bf16(qf0, load_b_frag(kb + (size_t)(j + 16) * 64, 64, lane), s1);
    s1 = wmma_bf16(qf1, load_b_frag(kb + (size_t)(j + 16) * 64 + 32, 64, lane), s1);

    // fetch V fragments early: overlap with softmax + LDS transpose
    const v16bf vf0 = load_b_frag(vb + (size_t)(0 * 16) * 2048 + j, 2048, lane);
    const v16bf vf1 = load_b_frag(vb + (size_t)(1 * 16) * 2048 + j, 2048, lane);
    const v16bf vf2 = load_b_frag(vb + (size_t)(2 * 16) * 2048 + j, 2048, lane);
    const v16bf vf3 = load_b_frag(vb + (size_t)(3 * 16) * 2048 + j, 2048, lane);

    float p0[8], p1[8], red[8];
#pragma unroll
    for (int r = 0; r < 8; ++r) {
      p0[r] = s0[r] * scale;
      p1[r] = s1[r] * scale;
      red[r] = fmaxf(p0[r], p1[r]);
    }
    // row-max across the 16 lanes of each half (rows live per-half)
#pragma unroll
    for (int mk = 8; mk >= 1; mk >>= 1)
#pragma unroll
      for (int r = 0; r < 8; ++r)
        red[r] = fmaxf(red[r], __shfl_xor(red[r], mk, 32));

    float alpha[8];
#pragma unroll
    for (int r = 0; r < 8; ++r) {
      const float mnew = fmaxf(mrow[r], red[r]);
      alpha[r] = __expf(mrow[r] - mnew);
      mrow[r] = mnew;
      p0[r] = __expf(p0[r] - mnew);
      p1[r] = __expf(p1[r] - mnew);
      red[r] = p0[r] + p1[r];
    }
#pragma unroll
    for (int mk = 8; mk >= 1; mk >>= 1)
#pragma unroll
      for (int r = 0; r < 8; ++r) red[r] += __shfl_xor(red[r], mk, 32);
#pragma unroll
    for (int r = 0; r < 8; ++r) lrow[r] = lrow[r] * alpha[r] + red[r];
#pragma unroll
    for (int t = 0; t < 4; ++t)
#pragma unroll
      for (int r = 0; r < 8; ++r) o[t][r] *= alpha[r];

    // P (C-layout) -> LDS row-major [16][32] -> A fragment
#pragma unroll
    for (int r = 0; r < 8; ++r) {
      const int row = r + half * 8;
      ps[row * 32 + col] = (bf16)p0[r];
      ps[row * 32 + col + 16] = (bf16)p1[r];
    }
    const v16bf pf = load_a_frag(ps, 32, lane);

    o[0] = wmma_bf16(pf, vf0, o[0]);
    o[1] = wmma_bf16(pf, vf1, o[1]);
    o[2] = wmma_bf16(pf, vf2, o[2]);
    o[3] = wmma_bf16(pf, vf3, o[3]);
  }

  float* ob = O + (bh * 2048 + q0) * 64;
#pragma unroll
  for (int r = 0; r < 8; ++r) {
    const float inv = 1.0f / lrow[r];
    const int row = r + half * 8;
#pragma unroll
    for (int t = 0; t < 4; ++t) ob[row * 64 + t * 16 + col] = o[t][r] * inv;
  }
}

// ---------------------------------------------------------------------------
// out = resid + LayerNorm(t) * g + be   over rows of 1024 (one block per row)
// Optionally also writes bf16 copy (FFN input).
// ---------------------------------------------------------------------------
__global__ __launch_bounds__(256) void ln_residual(
    const float* __restrict__ t, const float* __restrict__ resid,
    const float* __restrict__ g, const float* __restrict__ be,
    float* __restrict__ outf, bf16* __restrict__ outb) {
  const int row = blockIdx.x;
  const size_t base = (size_t)row * 1024 + threadIdx.x * 4;
  const float4 v = *(const float4*)(t + base);
  float s = v.x + v.y + v.z + v.w;
  float q = v.x * v.x + v.y * v.y + v.z * v.z + v.w * v.w;
#pragma unroll
  for (int mk = 16; mk >= 1; mk >>= 1) {
    s += __shfl_xor(s, mk, 32);
    q += __shfl_xor(q, mk, 32);
  }
  __shared__ float rs[8], rq[8];
  if ((threadIdx.x & 31) == 0) {
    rs[threadIdx.x >> 5] = s;
    rq[threadIdx.x >> 5] = q;
  }
  __syncthreads();
  float S = 0.f, QQ = 0.f;
#pragma unroll
  for (int i = 0; i < 8; ++i) { S += rs[i]; QQ += rq[i]; }
  const float mean = S * (1.0f / 1024.0f);
  const float var = QQ * (1.0f / 1024.0f) - mean * mean;
  const float rstd = rsqrtf(var + 1e-5f);

  const float4 r = *(const float4*)(resid + base);
  const int d = threadIdx.x * 4;
  const float o0 = r.x + (v.x - mean) * rstd * g[d + 0] + be[d + 0];
  const float o1 = r.y + (v.y - mean) * rstd * g[d + 1] + be[d + 1];
  const float o2 = r.z + (v.z - mean) * rstd * g[d + 2] + be[d + 2];
  const float o3 = r.w + (v.w - mean) * rstd * g[d + 3] + be[d + 3];
  *(float4*)(outf + base) = make_float4(o0, o1, o2, o3);
  if (outb) {
    outb[base + 0] = (bf16)o0;
    outb[base + 1] = (bf16)o1;
    outb[base + 2] = (bf16)o2;
    outb[base + 3] = (bf16)o3;
  }
}

// ---------------------------------------------------------------------------
extern "C" void kernel_launch(void* const* d_in, const int* in_sizes, int n_in,
                              void* d_out, int out_size, void* d_ws,
                              size_t ws_size, hipStream_t stream) {
  (void)in_sizes; (void)n_in; (void)out_size; (void)ws_size;
  const float* x   = (const float*)d_in[0];
  const float* ctx = (const float*)d_in[1];
  const float* Wq  = (const float*)d_in[2];
  const float* bq  = (const float*)d_in[3];
  const float* Wk  = (const float*)d_in[4];
  const float* bk  = (const float*)d_in[5];
  const float* Wv  = (const float*)d_in[6];
  const float* bv  = (const float*)d_in[7];
  const float* W1  = (const float*)d_in[8];
  const float* b1  = (const float*)d_in[9];
  const float* W2  = (const float*)d_in[10];
  const float* b2  = (const float*)d_in[11];
  const float* g1  = (const float*)d_in[12];
  const float* be1 = (const float*)d_in[13];
  const float* g2  = (const float*)d_in[14];
  const float* be2 = (const float*)d_in[15];
  float* out = (float*)d_out;

  char* w = (char*)d_ws;
  const size_t MB = (size_t)1 << 20;
  bf16*  xb    = (bf16*)(w + 0 * MB);     // 8 MiB  x bf16        [4096,1024]
  bf16*  cb    = (bf16*)(w + 8 * MB);     // 8 MiB  context bf16  [4096,1024]
  bf16*  Wqb   = (bf16*)(w + 16 * MB);    // 2 MiB
  bf16*  Wkb   = (bf16*)(w + 18 * MB);    // 2 MiB
  bf16*  Wvb   = (bf16*)(w + 20 * MB);    // 2 MiB
  bf16*  W1b   = (bf16*)(w + 22 * MB);    // 8 MiB  [4096,1024]
  bf16*  W2b   = (bf16*)(w + 30 * MB);    // 8 MiB  [1024,4096]
  bf16*  qb    = (bf16*)(w + 38 * MB);    // 8 MiB  [B,H,S,64]
  bf16*  kbuf  = (bf16*)(w + 46 * MB);    // 8 MiB  [B,H,K,64]
  bf16*  vtb   = (bf16*)(w + 54 * MB);    // 8 MiB  [B,H,64,K]
  float* ctxO  = (float*)(w + 62 * MB);   // 16 MiB [B,H,S,64] f32 (flat = reshape)
  float* out1f = (float*)(w + 78 * MB);   // 16 MiB [4096,1024] f32
  bf16*  out1b = (bf16*)(w + 94 * MB);    // 8 MiB
  bf16*  hb    = (bf16*)(w + 102 * MB);   // 32 MiB [4096,4096] bf16
  float* t2    = (float*)(w + 134 * MB);  // 16 MiB [4096,1024] f32

  // 1) f32 -> bf16 conversions
  cvt_bf16<<<dim3(2048), 256, 0, stream>>>(x, xb, 4194304);
  cvt_bf16<<<dim3(2048), 256, 0, stream>>>(ctx, cb, 4194304);
  cvt_bf16<<<dim3(1024), 256, 0, stream>>>(Wq, Wqb, 1048576);
  cvt_bf16<<<dim3(1024), 256, 0, stream>>>(Wk, Wkb, 1048576);
  cvt_bf16<<<dim3(1024), 256, 0, stream>>>(Wv, Wvb, 1048576);
  cvt_bf16<<<dim3(2048), 256, 0, stream>>>(W1, W1b, 4194304);
  cvt_bf16<<<dim3(2048), 256, 0, stream>>>(W2, W2b, 4194304);

  // 2) Q/K/V projections (M=4096, N=1024, K=1024)
  gemm_bf16_wmma<<<dim3(16, 32), 256, 0, stream>>>(xb, Wqb, bq, nullptr, qb,
                                                   4096, 1024, 1024, 2, 2048);
  gemm_bf16_wmma<<<dim3(16, 32), 256, 0, stream>>>(cb, Wkb, bk, nullptr, kbuf,
                                                   4096, 1024, 1024, 2, 2048);
  gemm_bf16_wmma<<<dim3(16, 32), 256, 0, stream>>>(cb, Wvb, bv, nullptr, vtb,
                                                   4096, 1024, 1024, 3, 2048);

  // 3) flash attention (grid: 32 q-tiles-of-4-waves x H x B)
  attn_wmma<<<dim3(32, 16, 2), ATTN_WPB * 32, 0, stream>>>(qb, kbuf, vtb, ctxO);

  // 4) out1 = x + LN(ctx_flat)
  ln_residual<<<dim3(4096), 256, 0, stream>>>(ctxO, x, g1, be1, out1f, out1b);

  // 5) FFN: h = gelu(out1 @ W1^T + b1)    (M=4096, N=4096, K=1024)
  gemm_bf16_wmma<<<dim3(64, 32), 256, 0, stream>>>(out1b, W1b, b1, nullptr, hb,
                                                   4096, 4096, 1024, 1, 1);
  //    t2 = h @ W2^T + b2                 (M=4096, N=1024, K=4096)
  gemm_bf16_wmma<<<dim3(16, 32), 256, 0, stream>>>(hb, W2b, b2, t2, nullptr,
                                                   4096, 1024, 4096, 0, 1);

  // 6) out = out1 + LN(t2)
  ln_residual<<<dim3(4096), 256, 0, stream>>>(t2, out1f, g2, be2, out, nullptr);
}
